// Sboxblock_33655363732243
// MI455X (gfx1250) — matrix operand order repaired and checked by reading, not compiled, not run on hardware
//
#include <hip/hip_runtime.h>
#include <hip/hip_bf16.h>

// 3D shifted-window attention block (Swin-style) for MI455X / gfx1250.
// All GEMMs run on v_wmma_f32_16x16x32_f16 (f16 in, fp32 accumulate).
// Activations staged in f16 to halve HBM traffic (workload is ~56 GFLOP vs
// ~400MB traffic => roughly bandwidth balanced at 23.3 TB/s).

#define BATCH 4
#define RDIM  32
#define SBOX  4
#define CDIM  128
#define HEADS 4
#define HD    32
#define MLPD  512
#define NTOK  64                         // tokens per window
#define MTOT  (BATCH * RDIM * RDIM * RDIM) // 131072 tokens

typedef _Float16 v8h  __attribute__((ext_vector_type(8)));
typedef _Float16 v16h __attribute__((ext_vector_type(16)));
typedef float    v8f  __attribute__((ext_vector_type(8)));

__device__ __forceinline__ v16h cat8(v8h lo, v8h hi) {
  return __builtin_shufflevector(lo, hi, 0, 1, 2, 3, 4, 5, 6, 7,
                                 8, 9, 10, 11, 12, 13, 14, 15);
}

// A-fragment for v_wmma_f32_16x16x32_f16 (ISA 7.12.2, 16-bit A 16x32):
// lane-half h: v[0..15] = K = h*16 + {0..15} (matches contiguous B fragments).
// p = &A[row][kbase] (row-major A).
__device__ __forceinline__ v16h load_a_frag(const _Float16* p, int half) {
  v8h lo = *(const v8h*)(p + half * 16);
  v8h hi = *(const v8h*)(p + half * 16 + 8);
  return cat8(lo, hi);
}

// One wave computes a 16x64 fp32 tile of A(16xK) @ B(64xK)^T.
// A row-major [16][K] (lda), B row-major [64][K] (ldb)  == KxN column-major,
// which gives the contiguous B-fragment order used here: lane n holds column n,
// 16 consecutive K starting at half*16.
template <int K>
__device__ __forceinline__ void wave_gemm_16x64(const _Float16* __restrict__ A, int lda,
                                                const _Float16* __restrict__ B, int ldb,
                                                int lane, v8f acc[4]) {
  const int nl = lane & 15;
  const int half = lane >> 4;
  const _Float16* arow = A + nl * lda;
#pragma unroll
  for (int k = 0; k < K; k += 32) {
    v16h a = load_a_frag(arow + k, half);
#pragma unroll
    for (int j = 0; j < 4; ++j) {
      v16h b = *(const v16h*)(B + (j * 16 + nl) * ldb + k + half * 16);
      acc[j] = __builtin_amdgcn_wmma_f32_16x16x32_f16(
          false, a, false, b, (short)0, acc[j], false, false);
    }
  }
}

// window-token index o -> source/destination global token index
// (roll(-SHIFT) + box_partition on the way in; its exact inverse on the way out)
__device__ __forceinline__ int src_token(int o) {
  int win = o >> 6, n = o & 63;
  int b = win >> 9, wi = win & 511;
  int wz = wi >> 6, wy = (wi >> 3) & 7, wx = wi & 7;
  int iz = n >> 4, iy = (n >> 2) & 3, ix = n & 3;
  int z = (wz * 4 + iz + 2) & 31;
  int y = (wy * 4 + iy + 2) & 31;
  int x = (wx * 4 + ix + 2) & 31;
  return ((b * 32 + z) * 32 + y) * 32 + x;
}

__global__ void f32_to_f16_kernel(const float* __restrict__ in,
                                  _Float16* __restrict__ out, int n) {
  int i = blockIdx.x * blockDim.x + threadIdx.x;
  if (i < n) out[i] = (_Float16)in[i];
}

// LayerNorm over C=128; one wave per token (4 channels/lane, float4 loads).
// shifted!=0: gather through roll/partition mapping (output in window order).
__global__ void ln_to_f16_kernel(const float* __restrict__ in,
                                 const float* __restrict__ gma,
                                 const float* __restrict__ bta,
                                 _Float16* __restrict__ out, int shifted) {
  int o = blockIdx.x * 8 + (threadIdx.x >> 5);
  int lane = threadIdx.x & 31;
  int t = shifted ? src_token(o) : o;
  const float4 v = *(const float4*)(in + (size_t)t * CDIM + lane * 4);
  float s = v.x + v.y + v.z + v.w;
  float q = v.x * v.x + v.y * v.y + v.z * v.z + v.w * v.w;
#pragma unroll
  for (int m = 1; m < 32; m <<= 1) {
    s += __shfl_xor(s, m, 32);
    q += __shfl_xor(q, m, 32);
  }
  float mean = s * (1.0f / CDIM);
  float var = q * (1.0f / CDIM) - mean * mean;
  float inv = rsqrtf(var + 1e-5f);
  float vv[4] = {v.x, v.y, v.z, v.w};
#pragma unroll
  for (int i = 0; i < 4; ++i) {
    int ch = lane * 4 + i;
    out[(size_t)o * CDIM + ch] = (_Float16)((vv[i] - mean) * inv * gma[ch] + bta[ch]);
  }
}

// QKV projection: [M,128] @ [384,128]^T, epilogue scatters into
// q (pre-scaled by 1/sqrt(32)), k, and v stored transposed per head.
__global__ void qkv_kernel(const _Float16* __restrict__ xw,
                           const _Float16* __restrict__ w,
                           _Float16* __restrict__ qb, _Float16* __restrict__ kb,
                           _Float16* __restrict__ vT) {
  const int wave = (blockIdx.x * blockDim.x + threadIdx.x) >> 5;
  const int lane = threadIdx.x & 31;
  const int mt = wave / 6;
  const int nt = wave % 6;
  v8f acc[4] = {};
  wave_gemm_16x64<CDIM>(xw + (size_t)mt * 16 * CDIM, CDIM,
                        w + (size_t)nt * 64 * CDIM, CDIM, lane, acc);
  const int nl = lane & 15, half = lane >> 4;
#pragma unroll
  for (int j = 0; j < 4; ++j) {
    int col = nt * 64 + j * 16 + nl;   // 0..383
    int which = col >> 7;              // 0=q 1=k 2=v
    int head = (col >> 5) & 3;
    int d = col & 31;
#pragma unroll
    for (int r = 0; r < 8; ++r) {
      int o = mt * 16 + r + 8 * half;  // window-token index
      int win = o >> 6, n = o & 63;
      size_t wh = (size_t)(win * HEADS + head);
      float val = acc[j][r];
      if (which == 0)
        qb[(wh * NTOK + n) * HD + d] = (_Float16)(val * 0.17677669529663687f);
      else if (which == 1)
        kb[(wh * NTOK + n) * HD + d] = (_Float16)val;
      else
        vT[(wh * HD + d) * NTOK + n] = (_Float16)val;  // transposed for PV GEMM
    }
  }
}

// One block per (window, head); 4 waves, each owning a 16-row strip.
__global__ void attn_kernel(const _Float16* __restrict__ qb,
                            const _Float16* __restrict__ kb,
                            const _Float16* __restrict__ vT,
                            const float* __restrict__ rpb,
                            const int* __restrict__ relidx,
                            const float* __restrict__ amask,
                            _Float16* __restrict__ xa) {
  __shared__ __align__(64) _Float16 P[4][16 * 64];  // per-wave softmax strip
  const int wh = blockIdx.x;
  const int win = wh >> 2;
  const int head = wh & 3;
  const int wi = win & 511;  // mask index
  const int wv = threadIdx.x >> 5;
  const int lane = threadIdx.x & 31;
  const int nl = lane & 15;
  const int half = lane >> 4;

  const _Float16* q = qb + (size_t)wh * (NTOK * HD);
  const _Float16* k = kb + (size_t)wh * (NTOK * HD);
  const _Float16* v = vT + (size_t)wh * (HD * NTOK);

  // S = q_strip(16x32) @ k^T : 4 WMMA (K=32 == full head dim)
  v8f s[4] = {};
  {
    v16h a = load_a_frag(q + (wv * 16 + nl) * HD, half);
#pragma unroll
    for (int j = 0; j < 4; ++j) {
      v16h b = *(const v16h*)(k + (j * 16 + nl) * HD + half * 16);
      s[j] = __builtin_amdgcn_wmma_f32_16x16x32_f16(
          false, a, false, b, (short)0, s[j], false, false);
    }
  }
  // + relative position bias + shifted-window mask  (C layout: (M=r+8*half, N=nl))
#pragma unroll
  for (int j = 0; j < 4; ++j) {
    int m = j * 16 + nl;
#pragma unroll
    for (int r = 0; r < 8; ++r) {
      int n = wv * 16 + r + 8 * half;
      s[j][r] += rpb[relidx[n * 64 + m] * HEADS + head] +
                 amask[((size_t)wi * 64 + n) * 64 + m];
    }
  }
  // row softmax: each row's 64 values live in slot r across one 16-lane half
  // (x 4 tiles); xor-shuffles 1,2,4,8 stay inside the half on wave32.
#pragma unroll
  for (int r = 0; r < 8; ++r) {
    float mx = fmaxf(fmaxf(s[0][r], s[1][r]), fmaxf(s[2][r], s[3][r]));
#pragma unroll
    for (int msk = 1; msk < 16; msk <<= 1) mx = fmaxf(mx, __shfl_xor(mx, msk, 32));
    float sum = 0.f;
#pragma unroll
    for (int j = 0; j < 4; ++j) {
      float e = __expf(s[j][r] - mx);
      s[j][r] = e;
      sum += e;
    }
#pragma unroll
    for (int msk = 1; msk < 16; msk <<= 1) sum += __shfl_xor(sum, msk, 32);
    float inv = 1.0f / sum;
#pragma unroll
    for (int j = 0; j < 4; ++j) s[j][r] *= inv;
  }
  // stage P row-major in LDS to re-layout from C-fragment into A-fragment
#pragma unroll
  for (int j = 0; j < 4; ++j)
#pragma unroll
    for (int r = 0; r < 8; ++r)
      P[wv][(r + 8 * half) * 64 + j * 16 + nl] = (_Float16)s[j][r];
  // O = P(16x64) @ V(64x32), B fragments contiguous thanks to vT layout
  v8f o[2] = {};
#pragma unroll
  for (int kc = 0; kc < 64; kc += 32) {
    v16h a = load_a_frag(&P[wv][nl * 64 + kc], half);
#pragma unroll
    for (int jo = 0; jo < 2; ++jo) {
      v16h b = *(const v16h*)(v + (jo * 16 + nl) * 64 + kc + half * 16);
      o[jo] = __builtin_amdgcn_wmma_f32_16x16x32_f16(
          false, a, false, b, (short)0, o[jo], false, false);
    }
  }
#pragma unroll
  for (int jo = 0; jo < 2; ++jo)
#pragma unroll
    for (int r = 0; r < 8; ++r) {
      int n = wv * 16 + r + 8 * half;
      int ch = head * HD + jo * 16 + nl;
      xa[((size_t)win * NTOK + n) * CDIM + ch] = (_Float16)o[jo][r];
    }
}

// proj GEMM + bias, fused with box_reverse + roll-back + residual (fp32 x1)
__global__ void proj_kernel(const _Float16* __restrict__ xa,
                            const _Float16* __restrict__ w,
                            const float* __restrict__ bias,
                            const float* __restrict__ x0,
                            float* __restrict__ x1) {
  const int wave = (blockIdx.x * blockDim.x + threadIdx.x) >> 5;
  const int lane = threadIdx.x & 31;
  const int mt = wave >> 1;
  const int nt = wave & 1;
  v8f acc[4] = {};
  wave_gemm_16x64<CDIM>(xa + (size_t)mt * 16 * CDIM, CDIM,
                        w + (size_t)nt * 64 * CDIM, CDIM, lane, acc);
  const int nl = lane & 15, half = lane >> 4;
#pragma unroll
  for (int j = 0; j < 4; ++j) {
    int col = nt * 64 + j * 16 + nl;
    float bv = bias[col];
#pragma unroll
    for (int r = 0; r < 8; ++r) {
      int o = mt * 16 + r + 8 * half;
      int t = src_token(o);  // inverse partition+roll
      size_t idx = (size_t)t * CDIM + col;
      x1[idx] = x0[idx] + 0.5f * (acc[j][r] + bv);
    }
  }
}

// MLP layer 1: [M,128]@[512,128]^T + b1, exact GELU, f16 out
__global__ void mlp1_kernel(const _Float16* __restrict__ hin,
                            const _Float16* __restrict__ w,
                            const float* __restrict__ bias,
                            _Float16* __restrict__ g) {
  const int wave = (blockIdx.x * blockDim.x + threadIdx.x) >> 5;
  const int lane = threadIdx.x & 31;
  const int mt = wave >> 3;
  const int nt = wave & 7;
  v8f acc[4] = {};
  wave_gemm_16x64<CDIM>(hin + (size_t)mt * 16 * CDIM, CDIM,
                        w + (size_t)nt * 64 * CDIM, CDIM, lane, acc);
  const int nl = lane & 15, half = lane >> 4;
#pragma unroll
  for (int j = 0; j < 4; ++j) {
    int col = nt * 64 + j * 16 + nl;
    float bv = bias[col];
#pragma unroll
    for (int r = 0; r < 8; ++r) {
      int t = mt * 16 + r + 8 * half;
      float u = acc[j][r] + bv;
      float ge = 0.5f * u * (1.0f + erff(u * 0.70710678118654752f));
      g[(size_t)t * MLPD + col] = (_Float16)ge;
    }
  }
}

// MLP layer 2: [M,512]@[128,512]^T + b2, residual, fp32 out
__global__ void mlp2_kernel(const _Float16* __restrict__ g,
                            const _Float16* __restrict__ w,
                            const float* __restrict__ bias,
                            const float* __restrict__ x1,
                            float* __restrict__ out) {
  const int wave = (blockIdx.x * blockDim.x + threadIdx.x) >> 5;
  const int lane = threadIdx.x & 31;
  const int mt = wave >> 1;
  const int nt = wave & 1;
  v8f acc[4] = {};
  wave_gemm_16x64<MLPD>(g + (size_t)mt * 16 * MLPD, MLPD,
                        w + (size_t)nt * 64 * MLPD, MLPD, lane, acc);
  const int nl = lane & 15, half = lane >> 4;
#pragma unroll
  for (int j = 0; j < 4; ++j) {
    int col = nt * 64 + j * 16 + nl;
    float bv = bias[col];
#pragma unroll
    for (int r = 0; r < 8; ++r) {
      int t = mt * 16 + r + 8 * half;
      size_t idx = (size_t)t * CDIM + col;
      out[idx] = x1[idx] + 0.5f * (acc[j][r] + bv);
    }
  }
}

extern "C" void kernel_launch(void* const* d_in, const int* in_sizes, int n_in,
                              void* d_out, int out_size, void* d_ws, size_t ws_size,
                              hipStream_t stream) {
  (void)in_sizes; (void)n_in; (void)out_size; (void)ws_size;
  const float* x      = (const float*)d_in[0];
  const float* qkv_w  = (const float*)d_in[1];
  const float* rpb    = (const float*)d_in[2];
  const float* proj_w = (const float*)d_in[3];
  const float* proj_b = (const float*)d_in[4];
  const float* n1g    = (const float*)d_in[5];
  const float* n1b    = (const float*)d_in[6];
  const float* n2g    = (const float*)d_in[7];
  const float* n2b    = (const float*)d_in[8];
  const float* w1     = (const float*)d_in[9];
  const float* b1     = (const float*)d_in[10];
  const float* w2     = (const float*)d_in[11];
  const float* b2     = (const float*)d_in[12];
  const float* amask  = (const float*)d_in[13];
  const int*   relidx = (const int*)d_in[14];
  float* out = (float*)d_out;

  size_t off = 0;
  auto carve = [&](size_t bytes) {
    char* r = (char*)d_ws + off;
    off = (off + bytes + 255) & ~(size_t)255;
    return (void*)r;
  };
  _Float16* wqkvh = (_Float16*)carve((size_t)3 * CDIM * CDIM * 2);
  _Float16* wprjh = (_Float16*)carve((size_t)CDIM * CDIM * 2);
  _Float16* w1h   = (_Float16*)carve((size_t)MLPD * CDIM * 2);
  _Float16* w2h   = (_Float16*)carve((size_t)CDIM * MLPD * 2);
  _Float16* xw    = (_Float16*)carve((size_t)MTOT * CDIM * 2);
  _Float16* qbuf  = (_Float16*)carve((size_t)MTOT * CDIM * 2);
  _Float16* kbuf  = (_Float16*)carve((size_t)MTOT * CDIM * 2);
  _Float16* vbuf  = (_Float16*)carve((size_t)MTOT * CDIM * 2);
  _Float16* xatt  = (_Float16*)carve((size_t)MTOT * CDIM * 2);
  float*    x1    = (float*)carve((size_t)MTOT * CDIM * 4);
  _Float16* hin   = (_Float16*)carve((size_t)MTOT * CDIM * 2);
  _Float16* gbuf  = (_Float16*)carve((size_t)MTOT * MLPD * 2);

  // weights -> f16 (tiny)
  f32_to_f16_kernel<<<(3 * CDIM * CDIM + 255) / 256, 256, 0, stream>>>(qkv_w, wqkvh, 3 * CDIM * CDIM);
  f32_to_f16_kernel<<<(CDIM * CDIM + 255) / 256, 256, 0, stream>>>(proj_w, wprjh, CDIM * CDIM);
  f32_to_f16_kernel<<<(MLPD * CDIM + 255) / 256, 256, 0, stream>>>(w1, w1h, MLPD * CDIM);
  f32_to_f16_kernel<<<(CDIM * MLPD + 255) / 256, 256, 0, stream>>>(w2, w2h, CDIM * MLPD);

  // LN1 fused with roll(-2)^3 + box partition (gather into window order)
  ln_to_f16_kernel<<<MTOT / 8, 256, 0, stream>>>(x, n1g, n1b, xw, 1);
  // QKV: M=131072, N=384, K=128  -> 49152 wave-tiles, 8 waves/block
  qkv_kernel<<<6144, 256, 0, stream>>>(xw, wqkvh, qbuf, kbuf, vbuf);
  // attention: one block per (window, head)
  attn_kernel<<<2048 * HEADS, 128, 0, stream>>>(qbuf, kbuf, vbuf, rpb, relidx, amask, xatt);
  // proj + reverse-map + residual -> x1 (fp32)
  proj_kernel<<<2048, 256, 0, stream>>>(xatt, wprjh, proj_b, x, x1);
  // LN2 (identity mapping)
  ln_to_f16_kernel<<<MTOT / 8, 256, 0, stream>>>(x1, n2g, n2b, hin, 0);
  // MLP
  mlp1_kernel<<<8192, 256, 0, stream>>>(hin, w1h, b1, gbuf);
  mlp2_kernel<<<2048, 256, 0, stream>>>(gbuf, w2h, b2, x1, out);
}
